// Decoder_48387101557101
// MI455X (gfx1250) — matrix operand order repaired and checked by reading, not compiled, not run on hardware
//
#include <hip/hip_runtime.h>
#include <hip/hip_bf16.h>
#include <math.h>

typedef __attribute__((ext_vector_type(16))) _Float16 v16h;
typedef __attribute__((ext_vector_type(8)))  _Float16 h8;
typedef __attribute__((ext_vector_type(8)))  float    v8f;

union V16 { v16h v; h8 h[2]; };

#define IEPS 1e-5f

__device__ __forceinline__ int refl(int i, int n) {
  if (i < 0) i = -i;
  if (i >= n) i = 2 * n - 2 - i;
  return i;
}

// ---------------------------------------------------------------------------
// Implicit-GEMM conv, reflect pad, f16 WMMA, f32 accumulate.
// One wave computes a 16(out-ch) x 64(pixel) tile (4 WMMA accumulators).
// inP : packed activations [B][HW][Cin]  (f16, channels-last)
// wP  : packed weights     [Cout][KS*KS][Cin] (f16, channels-last)
// Per K-chunk: A = 2 x b128 loads (shared by 4 WMMAs), each B = 2 x b128.
// ---------------------------------------------------------------------------
__global__ __launch_bounds__(32) void conv_wmma_kernel(
    const _Float16* __restrict__ inP, const _Float16* __restrict__ wP,
    const float* __restrict__ bias, float* __restrict__ out,
    int Cin, int Cout, int H, int W, int KS, int pad)
{
  const int lane = threadIdx.x;
  const int half = lane >> 4;   // 0: lanes 0-15, 1: lanes 16-31
  const int l16  = lane & 15;
  const int HW   = H * W;
  const int oBase = blockIdx.x * 16;
  const int pBase = blockIdx.y * 64;
  const int b     = blockIdx.z;
  const int K2 = KS * KS;

  const _Float16* wrow = wP + (size_t)(oBase + l16) * K2 * Cin; // A row M=l16
  const _Float16* inb  = inP + (size_t)b * HW * Cin;

  int pj[4], yj[4], xj[4];
  #pragma unroll
  for (int j = 0; j < 4; ++j) {
    pj[j] = pBase + j * 16 + l16;   // B column N = l16 (sub-tile j)
    yj[j] = pj[j] / W;
    xj[j] = pj[j] % W;
  }

  v8f acc[4] = {};
  for (int ky = 0; ky < KS; ++ky) {
    for (int kx = 0; kx < KS; ++kx) {
      const _Float16* wt = wrow + (ky * KS + kx) * Cin;
      __builtin_prefetch(wt + Cin, 0, 0);   // next tap's weight row
      const _Float16* bp[4];
      #pragma unroll
      for (int j = 0; j < 4; ++j) {
        int yy = refl(yj[j] + ky - pad, H);
        int xx = refl(xj[j] + kx - pad, W);
        bp[j] = inb + (size_t)(yy * W + xx) * Cin;
      }
      for (int c0 = 0; c0 < Cin; c0 += 32) {
        // A: K = kbA+e (e<8) and kbA+16+(e-8)  -> two contiguous h8 loads
        const _Float16* ap = wt + c0 + half * 8;
        V16 a;
        a.h[0] = *reinterpret_cast<const h8*>(ap);
        a.h[1] = *reinterpret_cast<const h8*>(ap + 16);
        const int kb = c0 + half * 8;       // B: K = kb+e (e<8) and kb+16+(e-8)
        #pragma unroll
        for (int j = 0; j < 4; ++j) {
          V16 bo;
          bo.h[0] = *reinterpret_cast<const h8*>(bp[j] + kb);
          bo.h[1] = *reinterpret_cast<const h8*>(bp[j] + kb + 16);
          acc[j] = __builtin_amdgcn_wmma_f32_16x16x32_f16(
              false, a.v, false, bo.v, (short)0, acc[j], false, false);
        }
      }
    }
  }
  #pragma unroll
  for (int r = 0; r < 8; ++r) {
    int oc = oBase + half * 8 + r;          // D: M = half*8 + r
    float bb = bias[oc];
    #pragma unroll
    for (int j = 0; j < 4; ++j)
      out[((size_t)b * Cout + oc) * HW + pj[j]] = acc[j][r] + bb;
  }
}

// ---------------------------------------------------------------------------
// Dense WMMA GEMM for the deform-conv einsum:
//   out[b][o][p] = sum_kk w[o][kk] * val[b][p][kk]   (kk = c*9 + k)
// valB packed f16 [B][HW][CK] (K-contiguous), wP packed f16 [Cout][CK].
// ---------------------------------------------------------------------------
__global__ __launch_bounds__(32) void gemm_wmma_kernel(
    const _Float16* __restrict__ valB, const _Float16* __restrict__ wP,
    const float* __restrict__ bias, float* __restrict__ out,
    int CK, int Cout, int HW)
{
  const int lane = threadIdx.x;
  const int half = lane >> 4;
  const int l16  = lane & 15;
  const int oBase = blockIdx.x * 16;
  const int pBase = blockIdx.y * 64;
  const int b     = blockIdx.z;

  const _Float16* wrow = wP + (size_t)(oBase + l16) * CK;
  const _Float16* vb   = valB + (size_t)b * HW * CK;

  int pj[4];
  const _Float16* bp[4];
  #pragma unroll
  for (int j = 0; j < 4; ++j) {
    pj[j] = pBase + j * 16 + l16;
    bp[j] = vb + (size_t)pj[j] * CK;
  }

  v8f acc[4] = {};
  for (int c0 = 0; c0 < CK; c0 += 32) {
    const _Float16* ap = wrow + c0 + half * 8;
    V16 a;
    a.h[0] = *reinterpret_cast<const h8*>(ap);
    a.h[1] = *reinterpret_cast<const h8*>(ap + 16);
    const int kb = c0 + half * 8;
    #pragma unroll
    for (int j = 0; j < 4; ++j) {
      V16 bo;
      bo.h[0] = *reinterpret_cast<const h8*>(bp[j] + kb);
      bo.h[1] = *reinterpret_cast<const h8*>(bp[j] + kb + 16);
      acc[j] = __builtin_amdgcn_wmma_f32_16x16x32_f16(
          false, a.v, false, bo.v, (short)0, acc[j], false, false);
    }
  }
  #pragma unroll
  for (int r = 0; r < 8; ++r) {
    int oc = oBase + half * 8 + r;
    float bb = bias[oc];
    #pragma unroll
    for (int j = 0; j < 4; ++j)
      out[((size_t)b * Cout + oc) * HW + pj[j]] = acc[j][r] + bb;
  }
}

// ---------------------------------------------------------------------------
// Packing kernels (f32 NCHW -> f16 channels-last), coalesced f16 writes.
// ---------------------------------------------------------------------------
__global__ void pack_act_kernel(const float* __restrict__ in,
                                _Float16* __restrict__ out, int B, int C, int HW)
{
  size_t n = (size_t)B * C * HW;
  size_t i = (size_t)blockIdx.x * blockDim.x + threadIdx.x;
  if (i >= n) return;
  int c = (int)(i % C);
  int p = (int)((i / C) % HW);
  int b = (int)(i / ((size_t)C * HW));
  out[i] = (_Float16)in[((size_t)b * C + c) * HW + p];
}

// fused 2x nearest upsample + pack: in f32 [B][C][H][W] -> out f16 [B][4HW][C]
__global__ void upsample_pack_kernel(const float* __restrict__ in,
                                     _Float16* __restrict__ out,
                                     int B, int C, int H, int W)
{
  int W2 = 2 * W;
  size_t HW2 = (size_t)4 * H * W;
  size_t n = (size_t)B * C * HW2;
  size_t i = (size_t)blockIdx.x * blockDim.x + threadIdx.x;
  if (i >= n) return;
  int c = (int)(i % C);
  size_t t = i / C;
  int p2 = (int)(t % HW2);
  int b  = (int)(t / HW2);
  int y2 = p2 / W2, x2 = p2 % W2;
  out[i] = (_Float16)in[((size_t)b * C + c) * H * W + (size_t)(y2 >> 1) * W + (x2 >> 1)];
}

// OIHW f32 -> [O][K2][C] f16
__global__ void pack_wconv_kernel(const float* __restrict__ in,
                                  _Float16* __restrict__ out, int O, int C, int K2)
{
  size_t n = (size_t)O * C * K2;
  size_t i = (size_t)blockIdx.x * blockDim.x + threadIdx.x;
  if (i >= n) return;
  int c = (int)(i % C);
  int t = (int)((i / C) % K2);
  int o = (int)(i / ((size_t)C * K2));
  out[i] = (_Float16)in[((size_t)o * C + c) * K2 + t];
}

// plain f32 -> f16 convert (deform weights are already K-contiguous)
__global__ void pack_f16_kernel(const float* __restrict__ in,
                                _Float16* __restrict__ out, size_t n)
{
  size_t i = (size_t)blockIdx.x * blockDim.x + threadIdx.x;
  if (i < n) out[i] = (_Float16)in[i];
}

// ---------------------------------------------------------------------------
// Instance norm over one (b,c) plane per block; in-place on f32 data.
// mode 1: relu(inorm(x)); mode 2: residual + inorm(x).
// ---------------------------------------------------------------------------
__global__ __launch_bounds__(256) void instnorm_kernel(
    float* __restrict__ data, const float* __restrict__ residual, int HW, int mode)
{
  __shared__ float ss[256], sq[256];
  const size_t base = (size_t)blockIdx.x * HW;
  float s = 0.f, q = 0.f;
  for (int i = threadIdx.x; i < HW; i += 256) {
    float v = data[base + i];
    s += v; q += v * v;
  }
  ss[threadIdx.x] = s; sq[threadIdx.x] = q;
  __syncthreads();
  for (int st = 128; st > 0; st >>= 1) {
    if (threadIdx.x < st) {
      ss[threadIdx.x] += ss[threadIdx.x + st];
      sq[threadIdx.x] += sq[threadIdx.x + st];
    }
    __syncthreads();
  }
  float mean = ss[0] / (float)HW;
  float var  = sq[0] / (float)HW - mean * mean;
  float inv  = rsqrtf(var + IEPS);
  for (int i = threadIdx.x; i < HW; i += 256) {
    float v = (data[base + i] - mean) * inv;
    if (mode == 1)      v = fmaxf(v, 0.f);
    else if (mode == 2) v += residual[base + i];
    data[base + i] = v;
  }
}

// Channel-concat copy: src [B][Csrc][HW] -> dst [B][Cdst][HW] at channel coff
__global__ void copy_ch_kernel(const float* __restrict__ src,
                               float* __restrict__ dst,
                               int B, int Csrc, int Cdst, int coff, int HW)
{
  size_t n = (size_t)B * Csrc * HW;
  size_t i = (size_t)blockIdx.x * blockDim.x + threadIdx.x;
  if (i >= n) return;
  int p = (int)(i % HW);
  int c = (int)((i / HW) % Csrc);
  int b = (int)(i / ((size_t)Csrc * HW));
  dst[((size_t)b * Cdst + coff + c) * HW + p] = src[i];
}

// Direct VALU conv for small Cout (offset convs, final 3-ch conv).
__global__ void conv_direct_kernel(const float* __restrict__ in,
                                   const float* __restrict__ wgt,
                                   const float* __restrict__ bias,
                                   float* __restrict__ out,
                                   int B, int Cin, int Cout, int H, int W,
                                   int KS, int pad, int reflectPad, int actTanh)
{
  int HW = H * W;
  size_t n = (size_t)B * Cout * HW;
  size_t i = (size_t)blockIdx.x * blockDim.x + threadIdx.x;
  if (i >= n) return;
  int p = (int)(i % HW);
  int o = (int)((i / HW) % Cout);
  int b = (int)(i / ((size_t)Cout * HW));
  int y = p / W, x = p % W;
  float acc = bias[o];
  const float* inb = in + (size_t)b * Cin * HW;
  const float* wr  = wgt + (size_t)o * Cin * KS * KS;
  for (int c = 0; c < Cin; ++c) {
    const float* ic = inb + (size_t)c * HW;
    const float* wc = wr + c * KS * KS;
    for (int ky = 0; ky < KS; ++ky) {
      int yy = y + ky - pad;
      if (reflectPad) yy = refl(yy, H);
      bool yok = (yy >= 0) && (yy < H);
      for (int kx = 0; kx < KS; ++kx) {
        int xx = x + kx - pad;
        if (reflectPad) xx = refl(xx, W);
        bool ok = yok && (xx >= 0) && (xx < W);
        float v = ok ? ic[yy * W + xx] : 0.f;
        acc = fmaf(v, wc[ky * KS + kx], acc);
      }
    }
  }
  if (actTanh) acc = tanhf(acc);
  out[i] = acc;
}

// Deformable sampling: val[b][p][c*9+k] (f16) = mask * bilinear(feat).
// Thread per (b,k,p); loops over channels (coords shared across c).
__global__ void deform_gather_kernel(const float* __restrict__ feat,
                                     const float* __restrict__ om,
                                     _Float16* __restrict__ val,
                                     int B, int C, int H, int W)
{
  int HW = H * W;
  size_t n = (size_t)B * 9 * HW;
  size_t i = (size_t)blockIdx.x * blockDim.x + threadIdx.x;
  if (i >= n) return;
  int p = (int)(i % HW);
  int k = (int)((i / HW) % 9);
  int b = (int)(i / ((size_t)9 * HW));
  int y = p / W, x = p % W;
  const float* omb = om + (size_t)b * 27 * HW;
  float dy = omb[(size_t)k * HW + p];
  float dx = omb[(size_t)(9 + k) * HW + p];
  float mk = 1.f / (1.f + __expf(-omb[(size_t)(18 + k) * HW + p]));
  float py = (float)y + (float)(k / 3 - 1) + dy;
  float px = (float)x + (float)(k % 3 - 1) + dx;
  float y0f = floorf(py), x0f = floorf(px);
  int y0 = (int)y0f, x0 = (int)x0f;
  int y1 = y0 + 1,  x1 = x0 + 1;
  float wy = py - y0f, wx = px - x0f;
  float w00 = (1.f - wy) * (1.f - wx), w01 = (1.f - wy) * wx;
  float w10 = wy * (1.f - wx),         w11 = wy * wx;
  if (y0 < 0 || y0 >= H) { w00 = 0.f; w01 = 0.f; }
  if (y1 < 0 || y1 >= H) { w10 = 0.f; w11 = 0.f; }
  if (x0 < 0 || x0 >= W) { w00 = 0.f; w10 = 0.f; }
  if (x1 < 0 || x1 >= W) { w01 = 0.f; w11 = 0.f; }
  int cy0 = min(max(y0, 0), H - 1), cy1 = min(max(y1, 0), H - 1);
  int cx0 = min(max(x0, 0), W - 1), cx1 = min(max(x1, 0), W - 1);
  int i00 = cy0 * W + cx0, i01 = cy0 * W + cx1;
  int i10 = cy1 * W + cx0, i11 = cy1 * W + cx1;
  const float* fb = feat + (size_t)b * C * HW;
  const int CK = C * 9;
  _Float16* vb = val + ((size_t)b * HW + p) * CK + k;   // stride 9 along c
  for (int c = 0; c < C; ++c) {
    const float* fc = fb + (size_t)c * HW;
    float v = w00 * fc[i00] + w01 * fc[i01] + w10 * fc[i10] + w11 * fc[i11];
    vb[(size_t)c * 9] = (_Float16)(v * mk);
  }
}

// Sum of |om[:, 0:18, :, :]| accumulated into *accum via atomicAdd.
__global__ __launch_bounds__(256) void abssum_kernel(const float* __restrict__ om,
                                                     float* __restrict__ accum,
                                                     int B, int HW)
{
  __shared__ float ss[256];
  size_t n = (size_t)B * 18 * HW;
  float s = 0.f;
  for (size_t i = (size_t)blockIdx.x * blockDim.x + threadIdx.x; i < n;
       i += (size_t)gridDim.x * blockDim.x) {
    int p = (int)(i % HW);
    int c = (int)((i / HW) % 18);
    int b = (int)(i / ((size_t)18 * HW));
    s += fabsf(om[((size_t)b * 27 + c) * HW + p]);
  }
  ss[threadIdx.x] = s;
  __syncthreads();
  for (int st = 128; st > 0; st >>= 1) {
    if (threadIdx.x < st) ss[threadIdx.x] += ss[threadIdx.x + st];
    __syncthreads();
  }
  if (threadIdx.x == 0) atomicAdd(accum, ss[0]);
}

__global__ void init_scalars_kernel(float* s) { s[0] = 0.f; s[1] = 0.f; }

__global__ void finalize_kernel(const float* __restrict__ s,
                                float* __restrict__ out, float cnt1, float cnt2)
{
  out[0] = 0.5f * (s[0] / cnt1 + s[1] / cnt2);
}

// ---------------------------------------------------------------------------
extern "C" void kernel_launch(void* const* d_in, const int* in_sizes, int n_in,
                              void* d_out, int out_size, void* d_ws, size_t ws_size,
                              hipStream_t stream)
{
  (void)in_sizes; (void)n_in; (void)out_size; (void)ws_size;
  const float* x      = (const float*)d_in[0];
  const float* skip1  = (const float*)d_in[1];
  const float* skip2  = (const float*)d_in[2];
  const float* rb1_w1 = (const float*)d_in[3];
  const float* rb1_b1 = (const float*)d_in[4];
  const float* rb1_w2 = (const float*)d_in[5];
  const float* rb1_b2 = (const float*)d_in[6];
  const float* rb2_w1 = (const float*)d_in[7];
  const float* rb2_b1 = (const float*)d_in[8];
  const float* rb2_w2 = (const float*)d_in[9];
  const float* rb2_b2 = (const float*)d_in[10];
  const float* c1_w   = (const float*)d_in[11];
  const float* c1_b   = (const float*)d_in[12];
  const float* c2_w   = (const float*)d_in[13];
  const float* c2_b   = (const float*)d_in[14];
  const float* c3_w   = (const float*)d_in[15];
  const float* c3_b   = (const float*)d_in[16];
  const float* off2_w = (const float*)d_in[17];
  const float* off2_b = (const float*)d_in[18];
  const float* d2_w   = (const float*)d_in[19];
  const float* d2_b   = (const float*)d_in[20];
  const float* off1_w = (const float*)d_in[21];
  const float* off1_b = (const float*)d_in[22];
  const float* d1_w   = (const float*)d_in[23];
  const float* d1_b   = (const float*)d_in[24];

  // Workspace layout (lifetime-based reuse; peak ~233 MB).
  char* ws = (char*)d_ws;
  const size_t MB = 1ull << 20;
  float*    t_h   = (float*)(ws + 0);          // 4MB
  float*    t_a   = (float*)(ws + 4 * MB);     // 4MB
  float*    t_b   = (float*)(ws + 8 * MB);     // 4MB
  float*    c1out = (float*)(ws + 12 * MB);    // 8MB
  float*    dcat2 = (float*)(ws + 20 * MB);    // 16MB
  float*    dcat1 = (float*)(ws + 20 * MB);    // 32MB (after dcat2 dead)
  float*    om2   = (float*)(ws + 52 * MB);    // ~1.7MB
  float*    om1   = (float*)(ws + 52 * MB);    // ~6.8MB (after om2 dead)
  _Float16* val2  = (_Float16*)(ws + 60 * MB); // 36MiB f16
  _Float16* val1  = (_Float16*)(ws + 60 * MB); // 72MiB f16 -> ends at 132MB
  float*    cpre2 = (float*)(ws + 132 * MB);   // 8MB
  float*    cpre1 = (float*)(ws + 132 * MB);   // 16MB
  float*    cat2  = (float*)(ws + 148 * MB);   // 16MB
  float*    cat1  = (float*)(ws + 148 * MB);   // 32MB
  float*    c2out = (float*)(ws + 180 * MB);   // 16MB
  _Float16* pact  = (_Float16*)(ws + 196 * MB);// 32MiB (max: up2 packed)
  _Float16* pwgt  = (_Float16*)(ws + 228 * MB);// <=2MB (transient, reused)
  float*    scal  = (float*)(ws + 232 * MB);   // 2 floats

  float* outImg    = (float*)d_out;              // 4*3*128*128
  float* outScalar = outImg + 4 * 3 * 128 * 128; // offset_sum

  dim3 blk32(32), blk256(256);
  auto NB = [](size_t n) { return (unsigned)((n + 255) / 256); };
  init_scalars_kernel<<<1, 1, 0, stream>>>(scal);

  // ---- resblock 1 (256ch, 32x32; HW=1024 -> 16 pixel-tiles of 64) ----
  pack_act_kernel<<<NB(4ull*256*1024), blk256, 0, stream>>>(x, pact, 4, 256, 1024);
  pack_wconv_kernel<<<NB(256ull*256*9), blk256, 0, stream>>>(rb1_w1, pwgt, 256, 256, 9);
  conv_wmma_kernel<<<dim3(16, 16, 4), blk32, 0, stream>>>(pact, pwgt, rb1_b1, t_h, 256, 256, 32, 32, 3, 1);
  instnorm_kernel<<<4 * 256, blk256, 0, stream>>>(t_h, nullptr, 1024, 1);
  pack_act_kernel<<<NB(4ull*256*1024), blk256, 0, stream>>>(t_h, pact, 4, 256, 1024);
  pack_wconv_kernel<<<NB(256ull*256*9), blk256, 0, stream>>>(rb1_w2, pwgt, 256, 256, 9);
  conv_wmma_kernel<<<dim3(16, 16, 4), blk32, 0, stream>>>(pact, pwgt, rb1_b2, t_a, 256, 256, 32, 32, 3, 1);
  instnorm_kernel<<<4 * 256, blk256, 0, stream>>>(t_a, x, 1024, 2);
  // ---- resblock 2 ----
  pack_act_kernel<<<NB(4ull*256*1024), blk256, 0, stream>>>(t_a, pact, 4, 256, 1024);
  pack_wconv_kernel<<<NB(256ull*256*9), blk256, 0, stream>>>(rb2_w1, pwgt, 256, 256, 9);
  conv_wmma_kernel<<<dim3(16, 16, 4), blk32, 0, stream>>>(pact, pwgt, rb2_b1, t_h, 256, 256, 32, 32, 3, 1);
  instnorm_kernel<<<4 * 256, blk256, 0, stream>>>(t_h, nullptr, 1024, 1);
  pack_act_kernel<<<NB(4ull*256*1024), blk256, 0, stream>>>(t_h, pact, 4, 256, 1024);
  pack_wconv_kernel<<<NB(256ull*256*9), blk256, 0, stream>>>(rb2_w2, pwgt, 256, 256, 9);
  conv_wmma_kernel<<<dim3(16, 16, 4), blk32, 0, stream>>>(pact, pwgt, rb2_b2, t_b, 256, 256, 32, 32, 3, 1);
  instnorm_kernel<<<4 * 256, blk256, 0, stream>>>(t_b, t_a, 1024, 2);
  // ---- upsample(+pack) -> packed [4][4096][256], then c1 5x5 256->128 ----
  upsample_pack_kernel<<<NB(4ull*256*4096), blk256, 0, stream>>>(t_b, pact, 4, 256, 32, 32);
  pack_wconv_kernel<<<NB(128ull*256*25), blk256, 0, stream>>>(c1_w, pwgt, 128, 256, 25);
  conv_wmma_kernel<<<dim3(8, 64, 4), blk32, 0, stream>>>(pact, pwgt, c1_b, c1out, 256, 128, 64, 64, 5, 2);
  instnorm_kernel<<<4 * 128, blk256, 0, stream>>>(c1out, nullptr, 4096, 1);
  // ---- dcat2 = [c1out, skip2] ----
  copy_ch_kernel<<<NB(4ull*128*4096), blk256, 0, stream>>>(c1out, dcat2, 4, 128, 256, 0, 4096);
  copy_ch_kernel<<<NB(4ull*128*4096), blk256, 0, stream>>>(skip2, dcat2, 4, 128, 256, 128, 4096);
  // ---- offset conv 2 (zero pad) + |offset| sum ----
  conv_direct_kernel<<<NB(4ull*27*4096), blk256, 0, stream>>>(dcat2, off2_w, off2_b, om2, 4, 256, 27, 64, 64, 3, 1, 0, 0);
  abssum_kernel<<<256, blk256, 0, stream>>>(om2, scal + 1, 4, 4096);
  // ---- deform gather 2 + einsum GEMM (CK=1152) ----
  deform_gather_kernel<<<NB(4ull*9*4096), blk256, 0, stream>>>(skip2, om2, val2, 4, 128, 64, 64);
  pack_f16_kernel<<<NB(128ull*1152), blk256, 0, stream>>>(d2_w, pwgt, 128ull*1152);
  gemm_wmma_kernel<<<dim3(8, 64, 4), blk32, 0, stream>>>(val2, pwgt, d2_b, cpre2, 1152, 128, 4096);
  // ---- cat2 = [cpre2, c1out]; upsample(+pack) -> [4][16384][256] ----
  copy_ch_kernel<<<NB(4ull*128*4096), blk256, 0, stream>>>(cpre2, cat2, 4, 128, 256, 0, 4096);
  copy_ch_kernel<<<NB(4ull*128*4096), blk256, 0, stream>>>(c1out, cat2, 4, 128, 256, 128, 4096);
  upsample_pack_kernel<<<NB(4ull*256*16384), blk256, 0, stream>>>(cat2, pact, 4, 256, 64, 64);
  // ---- c2: 5x5 256->64 @128x128 ----
  pack_wconv_kernel<<<NB(64ull*256*25), blk256, 0, stream>>>(c2_w, pwgt, 64, 256, 25);
  conv_wmma_kernel<<<dim3(4, 256, 4), blk32, 0, stream>>>(pact, pwgt, c2_b, c2out, 256, 64, 128, 128, 5, 2);
  instnorm_kernel<<<4 * 64, blk256, 0, stream>>>(c2out, nullptr, 16384, 1);
  // ---- dcat1 = [c2out, skip1] ----
  copy_ch_kernel<<<NB(4ull*64*16384), blk256, 0, stream>>>(c2out, dcat1, 4, 64, 128, 0, 16384);
  copy_ch_kernel<<<NB(4ull*64*16384), blk256, 0, stream>>>(skip1, dcat1, 4, 64, 128, 64, 16384);
  // ---- offset conv 1 (zero pad) + |offset| sum ----
  conv_direct_kernel<<<NB(4ull*27*16384), blk256, 0, stream>>>(dcat1, off1_w, off1_b, om1, 4, 128, 27, 128, 128, 3, 1, 0, 0);
  abssum_kernel<<<256, blk256, 0, stream>>>(om1, scal + 0, 4, 16384);
  // ---- deform gather 1 + einsum GEMM (CK=576) ----
  deform_gather_kernel<<<NB(4ull*9*16384), blk256, 0, stream>>>(skip1, om1, val1, 4, 64, 128, 128);
  pack_f16_kernel<<<NB(64ull*576), blk256, 0, stream>>>(d1_w, pwgt, 64ull*576);
  gemm_wmma_kernel<<<dim3(4, 256, 4), blk32, 0, stream>>>(val1, pwgt, d1_b, cpre1, 576, 64, 16384);
  // ---- cat1 = [cpre1, c2out] ----
  copy_ch_kernel<<<NB(4ull*64*16384), blk256, 0, stream>>>(cpre1, cat1, 4, 64, 128, 0, 16384);
  copy_ch_kernel<<<NB(4ull*64*16384), blk256, 0, stream>>>(c2out, cat1, 4, 64, 128, 64, 16384);
  // ---- c3: 7x7 128->3 reflect + tanh -> output image ----
  conv_direct_kernel<<<NB(4ull*3*16384), blk256, 0, stream>>>(cat1, c3_w, c3_b, outImg, 4, 128, 3, 128, 128, 7, 3, 1, 1);
  // ---- offset_sum scalar ----
  finalize_kernel<<<1, 1, 0, stream>>>(scal, outScalar,
                                       4.f * 18 * 16384, 4.f * 18 * 4096);
}
